// LSTM_32882269618408
// MI455X (gfx1250) — matrix operand order repaired and checked by reading, not compiled, not run on hardware
//
#include <hip/hip_runtime.h>
#include <cstdint>

// ---------------------------------------------------------------------------
// LSTM on MI455X (gfx1250, wave32, WMMA + TDM).
//   Phase 1: xg[B*T,400] = x[B*T,1536] * W_ih^T + (b_ih+b_hh)   (bf16 WMMA,
//            W_ih pre-converted to bf16; x converted in-register)
//   Phase 2: persistent single-WGP scan over T=512. Per step the xg[t] slice
//            (128x400 f32, 205 KB, L2-resident) is DMA'd into LDS by the
//            Tensor Data Mover; gate GEMM via WMMA on LDS-resident bf16 h
//            (K padded 100->128) against bf16 W_hh; nonlinearities on VALU,
//            c-state in registers.
//   Phase 3: tiny FC fused into scan epilogue.
// ---------------------------------------------------------------------------

typedef __bf16 v16bf __attribute__((ext_vector_type(16)));
typedef __bf16 v8bf  __attribute__((ext_vector_type(8)));
typedef float  v8f   __attribute__((ext_vector_type(8)));
typedef uint32_t v4u __attribute__((ext_vector_type(4)));
typedef int32_t  v8i __attribute__((ext_vector_type(8)));
typedef int32_t  v4i __attribute__((ext_vector_type(4)));

#define LSTM_B   128
#define LSTM_T   512
#define LSTM_D   1536
#define LSTM_H   100
#define LSTM_G   400      // 4*H
#define LSTM_HP  128      // H padded to K multiple of 32
#define LSTM_NT  (LSTM_B * LSTM_T)   // 65536 rows

// ---------------- helpers ----------------

__device__ __forceinline__ v16bf cvt16(float4 a0, float4 a1, float4 a2, float4 a3) {
    v16bf r;
    r[0]  = (__bf16)a0.x; r[1]  = (__bf16)a0.y; r[2]  = (__bf16)a0.z; r[3]  = (__bf16)a0.w;
    r[4]  = (__bf16)a1.x; r[5]  = (__bf16)a1.y; r[6]  = (__bf16)a1.z; r[7]  = (__bf16)a1.w;
    r[8]  = (__bf16)a2.x; r[9]  = (__bf16)a2.y; r[10] = (__bf16)a2.z; r[11] = (__bf16)a2.w;
    r[12] = (__bf16)a3.x; r[13] = (__bf16)a3.y; r[14] = (__bf16)a3.z; r[15] = (__bf16)a3.w;
    return r;
}

__device__ __forceinline__ float sigmoid_f(float x) {
    return 1.0f / (1.0f + __expf(-x));     // saturates cleanly at +-inf
}
__device__ __forceinline__ float tanh_f(float x) {
    x = fminf(15.0f, fmaxf(-15.0f, x));
    float e = __expf(2.0f * x);
    return (e - 1.0f) / (e + 1.0f);
}

// TDM: DMA xg[t] slice (128 rows x 400 f32, row stride T*400 elems) -> LDS.
// Descriptor bitfields per CDNA5 ISA.
// This toolchain's builtin takes 6 args: (g0, g1, g2, g3, g_extra, cpol).
__device__ __forceinline__ void tdm_load_xg(const float* gptr, uint32_t lds_off) {
    union { v4u v; uint32_t w[4]; } g0;
    union { v8i v; int32_t  w[8]; } g1;
    uint64_t ga = (uint64_t)(uintptr_t)gptr;
    g0.w[0] = 1u;                                  // count=1 (valid), user mode
    g0.w[1] = lds_off;                             // lds_addr (bytes)
    g0.w[2] = (uint32_t)ga;                        // global_addr[31:0]
    g0.w[3] = (uint32_t)((ga >> 32) & 0x01FFFFFFu) // global_addr[56:32]
            | (2u << 30);                          // type=2 ("image")
    g1.w[0] = (int)(2u << 16);                     // data_size=2 -> 4 bytes
    g1.w[1] = (int)((uint32_t)LSTM_G << 16);       // tensor_dim0=400 (bits 79:48, lo16)
    g1.w[2] = (int)((uint32_t)LSTM_B << 16);       // dim0 hi=0 | tensor_dim1=128 (lo16)
    g1.w[3] = (int)((uint32_t)LSTM_G << 16);       // dim1 hi=0 | tile_dim0=400
    g1.w[4] = LSTM_B;                              // tile_dim1=128 | tile_dim2=0
    g1.w[5] = LSTM_T * LSTM_G;                     // tensor_dim0_stride = 204800 elems
    g1.w[6] = 0;
    g1.w[7] = 0;
    v4i z4 = (v4i){0, 0, 0, 0};
    v8i z8 = (v8i){0, 0, 0, 0, 0, 0, 0, 0};
    __builtin_amdgcn_tensor_load_to_lds(g0.v, g1.v, z4, z4, z8, 0);
}

// ---------------- kernel 1: prep (bias sum + bf16 weight conversions) ------

__global__ void lstm_prep(const float* __restrict__ Wih,
                          const float* __restrict__ Whh,
                          const float* __restrict__ bih,
                          const float* __restrict__ bhh,
                          __bf16* __restrict__ Wih_b,
                          __bf16* __restrict__ Whh_b,
                          float*  __restrict__ bias) {
    int i = blockIdx.x * blockDim.x + threadIdx.x;
    int stride = gridDim.x * blockDim.x;
    if (i < LSTM_G) bias[i] = bih[i] + bhh[i];
    for (int idx = i; idx < LSTM_G * LSTM_D; idx += stride)
        Wih_b[idx] = (__bf16)Wih[idx];
    for (int idx = i; idx < LSTM_G * LSTM_HP; idx += stride) {
        int g = idx >> 7;            // /128
        int k = idx & (LSTM_HP - 1);
        Whh_b[idx] = (k < LSTM_H) ? (__bf16)Whh[g * LSTM_H + k] : (__bf16)0.0f;
    }
}

// ---------------- kernel 2: xg = x * W_ih^T + bias -------------------------
// One wave per 16-row M-tile; each wave owns all 25 N-tiles (x streamed once
// from HBM; bf16 W_ih served from L2). K = 1536 -> 48 steps of 32 (bf16).

__global__ __launch_bounds__(256) void lstm_xg_gemm(
        const float*  __restrict__ x,
        const __bf16* __restrict__ Wih_b,
        const float*  __restrict__ bias,
        float* __restrict__ xg) {
    const int lane = threadIdx.x & 31;
    const int wave = threadIdx.x >> 5;
    const int mtile = blockIdx.x * 8 + wave;      // 4096 M-tiles
    const int half = lane >> 4;                   // 0 or 1
    const int lm   = lane & 15;

    const float* __restrict__ arow = x + (size_t)(mtile * 16 + lm) * LSTM_D;

    v8f acc[25];
#pragma unroll
    for (int n = 0; n < 25; ++n) acc[n] = (v8f){};

    for (int kk = 0; kk < LSTM_D / 32; ++kk) {
        const int kbase = kk * 32;
        // A fragment: row m = lane%16; chunks at 8*half and 8*half+16
        const int ka = kbase + 8 * half;
        float4 a0 = *(const float4*)(arow + ka);
        float4 a1 = *(const float4*)(arow + ka + 4);
        float4 a2 = *(const float4*)(arow + ka + 16);
        float4 a3 = *(const float4*)(arow + ka + 20);
        v16bf afrag = cvt16(a0, a1, a2, a3);

        const int kb = kbase + 8 * half;
#pragma unroll
        for (int n = 0; n < 25; ++n) {
            // B fragment: col n = lane%16; chunks at 8*half and 8*half+16
            const __bf16* __restrict__ brow =
                Wih_b + (size_t)(n * 16 + lm) * LSTM_D + kb;
            v8bf b0 = *(const v8bf*)(brow);
            v8bf b1 = *(const v8bf*)(brow + 16);
            v16bf bfrag = __builtin_shufflevector(b0, b1,
                0, 1, 2, 3, 4, 5, 6, 7, 8, 9, 10, 11, 12, 13, 14, 15);
            acc[n] = __builtin_amdgcn_wmma_f32_16x16x32_bf16(
                false, afrag, false, bfrag, (short)0, acc[n], false, false);
        }
    }

#pragma unroll
    for (int n = 0; n < 25; ++n) {
        const int col = n * 16 + lm;
        const float bc = bias[col];
#pragma unroll
        for (int r = 0; r < 8; ++r) {
            const int row = mtile * 16 + r + 8 * half;   // C/D layout
            xg[(size_t)row * LSTM_G + col] = acc[n][r] + bc;
        }
    }
}

// ---------------- kernel 3: persistent recurrent scan + FC -----------------
// Single workgroup, 1024 threads (32 waves). Per step:
//   TDM has pre-filled LDS gates with xg[:,t,:] (f32).
//   Phase A: 200 16x16 gate tiles, 4 x wmma(bf16,K=32) each; A from LDS h,
//            B from global bf16 W_hh; gates += acc (in-place LDS).
//   Phase B: 12800 (b,h) cells: i,f,g,o nonlinearities, c in registers,
//            new h written back to LDS as bf16. Then TDM for t+1 issued.

#define GATES_BYTES (LSTM_B * LSTM_G * 4)        // 204800
#define HBF_BYTES   (LSTM_B * LSTM_HP * 2)       // 32768
#define HFIN_BYTES  (LSTM_B * LSTM_H * 4)        // 51200
#define SCAN_SMEM   (GATES_BYTES + HBF_BYTES + HFIN_BYTES)

__global__ __launch_bounds__(1024) void lstm_scan(
        const float*  __restrict__ xg,
        const __bf16* __restrict__ Whh_b,
        const float*  __restrict__ Wfc,
        const float*  __restrict__ bfc,
        float* __restrict__ out) {
    extern __shared__ char smem[];
    float*  gates = (float*)smem;
    __bf16* hbf   = (__bf16*)(smem + GATES_BYTES);
    float*  hfin  = (float*)(smem + GATES_BYTES + HBF_BYTES);

    const int tid  = threadIdx.x;
    const int lane = tid & 31;
    const int wave = tid >> 5;
    const int half = lane >> 4;
    const int lm   = lane & 15;
    const uint32_t gates_lds = (uint32_t)(uintptr_t)gates;  // LDS byte offset

    // Kick off DMA of xg[:,0,:] while we zero-init state.
    if (wave == 0) tdm_load_xg(xg, gates_lds);

    for (int i = tid; i < (LSTM_B * LSTM_HP) / 2; i += 1024)
        ((uint32_t*)hbf)[i] = 0u;
    float creg[13];
#pragma unroll
    for (int j = 0; j < 13; ++j) creg[j] = 0.0f;

    for (int t = 0; t < LSTM_T; ++t) {
        if (wave == 0) __builtin_amdgcn_s_wait_tensorcnt((short)0);
        __syncthreads();   // gates now holds xg[:,t,:]; hbf from prev step

        // ---- Phase A: gates += h @ Whh^T ----
        for (int tile = wave; tile < 200; tile += 32) {
            const int mt = tile / 25;
            const int nt = tile - mt * 25;
            const int m   = mt * 16 + lm;        // A row (batch)
            const int col = nt * 16 + lm;        // B col (gate index)
            const __bf16* __restrict__ brow = Whh_b + (size_t)col * LSTM_HP;
            v8f acc = (v8f){};
#pragma unroll
            for (int kk = 0; kk < LSTM_HP / 32; ++kk) {
                const int ka = kk * 32 + 8 * half;
                v8bf alo = *(const v8bf*)(hbf + m * LSTM_HP + ka);
                v8bf ahi = *(const v8bf*)(hbf + m * LSTM_HP + ka + 16);
                v16bf afrag = __builtin_shufflevector(alo, ahi,
                    0, 1, 2, 3, 4, 5, 6, 7, 8, 9, 10, 11, 12, 13, 14, 15);
                const int kb = kk * 32 + 8 * half;
                v8bf b0 = *(const v8bf*)(brow + kb);
                v8bf b1 = *(const v8bf*)(brow + kb + 16);
                v16bf bfrag = __builtin_shufflevector(b0, b1,
                    0, 1, 2, 3, 4, 5, 6, 7, 8, 9, 10, 11, 12, 13, 14, 15);
                acc = __builtin_amdgcn_wmma_f32_16x16x32_bf16(
                    false, afrag, false, bfrag, (short)0, acc, false, false);
            }
#pragma unroll
            for (int r = 0; r < 8; ++r) {
                const int brow_i = mt * 16 + r + 8 * half;   // batch index
                gates[brow_i * LSTM_G + col] += acc[r];
            }
        }
        __syncthreads();

        // ---- Phase B: nonlinearities + state update ----
        int j = 0;
        for (int idx = tid; idx < LSTM_B * LSTM_H; idx += 1024, ++j) {
            const int b  = idx / LSTM_H;
            const int hh = idx - b * LSTM_H;
            const float* g4 = gates + b * LSTM_G;
            const float i_s = sigmoid_f(g4[hh]);
            const float f_s = sigmoid_f(g4[LSTM_H + hh]);
            const float g_t = tanh_f   (g4[2 * LSTM_H + hh]);
            const float o_s = sigmoid_f(g4[3 * LSTM_H + hh]);
            const float c = f_s * creg[j] + i_s * g_t;
            creg[j] = c;
            const float h = o_s * tanh_f(c);
            hbf[b * LSTM_HP + hh] = (__bf16)h;
            if (t == LSTM_T - 1) hfin[idx] = h;
        }
        __syncthreads();   // gates free; hbf complete

        // DMA next step's xg slice (overlaps loop backedge / wait ramp)
        if (wave == 0 && t + 1 < LSTM_T)
            tdm_load_xg(xg + (size_t)(t + 1) * LSTM_G, gates_lds);
    }

    // ---- FC: out[b,o] = h_T[b,:] . Wfc[o,:] + bfc[o] ----
    if (tid < LSTM_B * 4) {
        const int b = tid >> 2;
        const int o = tid & 3;
        const float* wr = Wfc + o * LSTM_H;
        const float* hr = hfin + b * LSTM_H;
        float s = bfc[o];
        for (int k = 0; k < LSTM_H; ++k) s += hr[k] * wr[k];
        out[b * 4 + o] = s;
    }
}

// ---------------- launch ----------------------------------------------------

extern "C" void kernel_launch(void* const* d_in, const int* in_sizes, int n_in,
                              void* d_out, int out_size, void* d_ws, size_t ws_size,
                              hipStream_t stream) {
    const float* x    = (const float*)d_in[0];
    const float* Wih  = (const float*)d_in[1];
    const float* Whh  = (const float*)d_in[2];
    const float* bih  = (const float*)d_in[3];
    const float* bhh  = (const float*)d_in[4];
    const float* Wfc  = (const float*)d_in[5];
    const float* bfc  = (const float*)d_in[6];
    float* out = (float*)d_out;

    char* ws = (char*)d_ws;
    size_t off = 0;
    float*  xg    = (float*)(ws + off);  off += (size_t)LSTM_NT * LSTM_G * 4;   // 100 MB
    __bf16* Whh_b = (__bf16*)(ws + off); off += (size_t)LSTM_G * LSTM_HP * 2;   // 100 KB
    __bf16* Wih_b = (__bf16*)(ws + off); off += (size_t)LSTM_G * LSTM_D * 2;    // 1.2 MB
    float*  bias  = (float*)(ws + off);  off += (size_t)LSTM_G * 4;             // 1.6 KB

    lstm_prep<<<128, 256, 0, stream>>>(Wih, Whh, bih, bhh, Wih_b, Whh_b, bias);
    lstm_xg_gemm<<<LSTM_NT / (16 * 8), 256, 0, stream>>>(x, Wih_b, bias, xg);

    (void)hipFuncSetAttribute((const void*)lstm_scan,
                              hipFuncAttributeMaxDynamicSharedMemorySize,
                              SCAN_SMEM);
    lstm_scan<<<1, 1024, SCAN_SMEM, stream>>>(xg, Whh_b, Wfc, bfc, out);
}
